// TransformerMHWarpLayer_25692494365452
// MI455X (gfx1250) — matrix operand order repaired and checked by reading, not compiled, not run on hardware
//
#include <hip/hip_runtime.h>
#include <hip/hip_bf16.h>
#include <math.h>

typedef __attribute__((ext_vector_type(16))) _Float16 v16h;
typedef __attribute__((ext_vector_type(8)))  _Float16 v8h;
typedef __attribute__((ext_vector_type(8)))  float    v8f;

// Sizes (fixed by the reference)
#define NQ   4096
#define NK   4096
#define BH   4
#define HD   64
#define POSK 384
#define EPSF 2.220446049250313e-16f
#define NEG_BIG -3.4e38f
// scores are cosines of unit vectors (|f| <= ~1), so softmax can use the
// fixed stabilizer max=1:  p = exp((f-1)/T) = 2^(f*S - S), S = 100*log2(e)
#define SCALE_LOG2E 144.26950408889634f

// d_out sections (floats)
#define OUT_BASE   0        // [8192][2][6]
#define SRCY_BASE  98304    // [4096][2][4]
#define SMAP_BASE  131072   // [4096][2][2]

// ---------------- fragment helpers (CDNA5 WMMA layouts, ISA 7.12.2) -------
// A 16x32 f16 (K order used here, same as B): lane l: M = l&15; half h = l>>4; elems 0..7 -> K = (h?16:0)+t,
// elems 8..15 -> K = (h?24:8)+t  (relative to kbase)
__device__ __forceinline__ v16h afrag_f16(const _Float16* row, int klo, int khi) {
  v8h lo = *(const v8h*)(row + klo);
  v8h hi = *(const v8h*)(row + khi);
  v16h a;
#pragma unroll
  for (int t = 0; t < 8; ++t) { a[t] = lo[t]; a[8 + t] = hi[t]; }
  return a;
}

// ---------------------------- k0: const fill ------------------------------
__global__ void k0_fill(const float* __restrict__ ref_ab, float* __restrict__ out,
                        float* __restrict__ sums) {
  int gid = blockIdx.x * 256 + threadIdx.x;
  if (gid < 8192) {                      // key rows of `out`
    int n = gid >> 1, b = gid & 1;
    float r0 = ref_ab[n * 4 + b * 2], r1 = ref_ab[n * 4 + b * 2 + 1];
    size_t ob = (size_t)n * 12 + b * 6;
    out[ob + 0] = r0; out[ob + 1] = r1; out[ob + 2] = r0; out[ob + 3] = r1;
    out[ob + 4] = 1.0f; out[ob + 5] = 1.0f;
  } else if (gid < 8192 + 512) {
    sums[gid - 8192] = 0.0f;
  }
}

// ------------------- k1: pos_e = pos @ pos_w.T + pos_b --------------------
// M=16384 (m = n*2+b), K=384, N=64.  Block: 8 waves -> 32 rows x 64 cols.
__global__ void __launch_bounds__(256) k1_pos_gemm(
    const float* __restrict__ pos, const float* __restrict__ pos_w,
    const float* __restrict__ pos_b, float* __restrict__ pos_e) {
  int t = threadIdx.x, lane = t & 31, wv = t >> 5;
  int half = lane >> 4, l15 = lane & 15;
  int ct = wv & 3;                               // col tile 0..3
  int rowbase = blockIdx.x * 32 + (wv >> 2) * 16;
  int klo = half ? 16 : 0, khi = half ? 24 : 8;
  int d = ct * 16 + l15;                         // output / B-matrix column
  const float* arow = pos + (size_t)(rowbase + l15) * POSK;
  const float* wrow = pos_w + (size_t)d * POSK + (half ? 16 : 0);
  v8f c = {};
  for (int kb = 0; kb < POSK; kb += 32) {
    v16h a, b;
#pragma unroll
    for (int tt = 0; tt < 8; ++tt) {
      a[tt]     = (_Float16)arow[kb + klo + tt];
      a[8 + tt] = (_Float16)arow[kb + khi + tt];
    }
#pragma unroll
    for (int tt = 0; tt < 16; ++tt) b[tt] = (_Float16)wrow[kb + tt];
    c = __builtin_amdgcn_wmma_f32_16x16x32_f16(false, a, false, b, (short)0, c,
                                               false, false);
  }
  float bias = pos_b[d];
#pragma unroll
  for (int v = 0; v < 8; ++v)
    pos_e[(size_t)(rowbase + v + 8 * half) * 64 + d] = c[v] + bias;
}

// --------- k2: theta/phi raw projections + column sums (for mean) ---------
// grid (32, bh=4, which=2); block 256 = 8 waves; 128 rows staged in LDS.
__global__ void __launch_bounds__(256) k2_proj(
    const float* __restrict__ src, const float* __restrict__ pos_e,
    const float* __restrict__ theta_w, const float* __restrict__ theta_b,
    const float* __restrict__ phi_w, const float* __restrict__ phi_b,
    float* __restrict__ theta_raw, float* __restrict__ phi_raw,
    float* __restrict__ sums) {
  __shared__ float tile[128 * 65];               // padded: 64 banks, stride 65
  int t = threadIdx.x;
  int bh = blockIdx.y;                           // 0..3, bh = b*2 + h
  int which = blockIdx.z;                        // 0: theta(frame1,+pos)  1: phi(frame0,-pos)
  int i0 = blockIdx.x * 128;
  int b = bh >> 1, h = bh & 1;
  int frame = which ? 0 : 1;
  // ---- stage q/k head tile, coalesced over HW (innermost of src) ----
  for (int r = 0; r < 32; ++r) {
    int flat = t + 256 * r;                      // 128 rows x 64 chans
    int e = flat >> 7, i = flat & 127;
    int cch = 64 * h + e;
    float val = src[(((size_t)(b * 128 + cch)) * 2 + frame) * 4096 + i0 + i];
    if (bh >= 2) {                               // q[:,B:2B]/k[:,B:2B] pos add
      int j = bh - 2;
      int npos = (which ? 0 : NK) + i0 + i;
      float pe = pos_e[((size_t)npos * 2 + j) * 64 + e];
      val += which ? -pe : pe;
    }
    tile[i * 65 + e] = val;
  }
  __syncthreads();
  int lane = t & 31, wv = t >> 5;
  int half = lane >> 4, l15 = lane & 15;
  int klo = half ? 16 : 0, khi = half ? 24 : 8;
  const float* W  = which ? phi_w : theta_w;
  const float* Bv = which ? phi_b : theta_b;
  float* Out      = which ? phi_raw : theta_raw;
  const float* lr = &tile[(wv * 16 + l15) * 65];
  v16h a0, a1;
#pragma unroll
  for (int tt = 0; tt < 8; ++tt) {
    a0[tt] = (_Float16)lr[klo + tt];      a0[8 + tt] = (_Float16)lr[khi + tt];
    a1[tt] = (_Float16)lr[32 + klo + tt]; a1[8 + tt] = (_Float16)lr[32 + khi + tt];
  }
#pragma unroll
  for (int ct = 0; ct < 4; ++ct) {
    int d = ct * 16 + l15;
    const float* wr = W + (size_t)d * 64 + (half ? 16 : 0);
    v16h b0, b1;
#pragma unroll
    for (int tt = 0; tt < 16; ++tt) {
      b0[tt] = (_Float16)wr[tt];
      b1[tt] = (_Float16)wr[32 + tt];
    }
    v8f c = {};
    c = __builtin_amdgcn_wmma_f32_16x16x32_f16(false, a0, false, b0, (short)0, c, false, false);
    c = __builtin_amdgcn_wmma_f32_16x16x32_f16(false, a1, false, b1, (short)0, c, false, false);
    float bias = Bv[d];
    float partial = 0.f;
#pragma unroll
    for (int v = 0; v < 8; ++v) {
      float val = c[v] + bias;
      partial += val;
      int row = i0 + wv * 16 + v + 8 * half;
      Out[((size_t)bh * 4096 + row) * 64 + d] = val;
    }
    atomicAdd(&sums[which * 256 + bh * 64 + d], partial);
  }
}

// ------- k3: mean-center per channel, L2-normalize per sample, -> f16 -----
__global__ void __launch_bounds__(256) k3_norm(
    const float* __restrict__ raw_t, const float* __restrict__ raw_p,
    const float* __restrict__ sums, _Float16* __restrict__ th,
    _Float16* __restrict__ ph) {
  int bh = blockIdx.y, which = blockIdx.z;
  int lane = threadIdx.x & 31, wv = threadIdx.x >> 5;
  int row = blockIdx.x * 8 + wv;
  const float* raw = which ? raw_p : raw_t;
  _Float16* outp = which ? ph : th;
  size_t base = ((size_t)bh * 4096 + row) * 64;
  float m0 = sums[which * 256 + bh * 64 + lane]      * (1.0f / 4096.0f);
  float m1 = sums[which * 256 + bh * 64 + lane + 32] * (1.0f / 4096.0f);
  float a  = raw[base + lane]      - m0;
  float bb = raw[base + lane + 32] - m1;
  float ss = a * a + bb * bb;
#pragma unroll
  for (int mask = 1; mask < 32; mask <<= 1) ss += __shfl_xor(ss, mask, 32);
  float inv = 1.0f / (sqrtf(ss) + EPSF);
  outp[base + lane]      = (_Float16)(a * inv);
  outp[base + lane + 32] = (_Float16)(bb * inv);
}

// -------------- k4: streaming scores + fixed-max softmax + V-accum --------
// Wave per 16-query tile; 256 key tiles.  Scores are cosines (<= 1), so the
// softmax stabilizer is the constant 1.0: no running-max rescaling, one
// v_exp per score.  Raw max still tracked (cheap fmax) for the sim output.
__global__ void __launch_bounds__(256) k4_attn(
    const _Float16* __restrict__ th, const _Float16* __restrict__ ph,
    const float* __restrict__ ref_ab, float* __restrict__ out) {
  int t = threadIdx.x, lane = t & 31, wv = t >> 5;
  int half = lane >> 4, l15 = lane & 15;
  int bh = blockIdx.y;
  int qt = blockIdx.x * 8 + wv;
  int klo = half ? 16 : 0, khi = half ? 24 : 8;
  const _Float16* arow = th + ((size_t)bh * 4096 + qt * 16 + l15) * 64;
  v16h a0 = afrag_f16(arow, klo, khi);
  v16h a1 = afrag_f16(arow + 32, klo, khi);
  int bref = bh & 1;                              // v[n,bh] = ref_ab[n, bh%2]
  float mm[8], ss[8], y0[8], y1[8];
#pragma unroll
  for (int v = 0; v < 8; ++v) { mm[v] = NEG_BIG; ss[v] = 0.f; y0[v] = 0.f; y1[v] = 0.f; }
  const _Float16* pbase = ph + (size_t)bh * 4096 * 64;
  int koff = half ? 16 : 0;
  for (int j = 0; j < 256; ++j) {
    int n = j * 16 + l15;                         // this lane's key column
    const _Float16* prow = pbase + (size_t)n * 64;
    if (j + 8 < 256) __builtin_prefetch(prow + 8 * 16 * 64, 0, 1);  // global_prefetch
    v16h b0 = *(const v16h*)(prow + koff);        // B frag: K contiguous
    v16h b1 = *(const v16h*)(prow + 32 + koff);
    v8f c = {};
    c = __builtin_amdgcn_wmma_f32_16x16x32_f16(false, a0, false, b0, (short)0, c, false, false);
    c = __builtin_amdgcn_wmma_f32_16x16x32_f16(false, a1, false, b1, (short)0, c, false, false);
    float va = ref_ab[(size_t)n * 4 + bref * 2 + 0];
    float vb = ref_ab[(size_t)n * 4 + bref * 2 + 1];
#pragma unroll
    for (int v = 0; v < 8; ++v) {                 // fixed-stabilizer softmax
      float val = c[v];
      mm[v] = fmaxf(mm[v], val);
      // native v_exp_f32 (2^x): p = 2^(val*S - S) = exp((val-1)/T)
      float p = __builtin_amdgcn_exp2f(fmaf(val, SCALE_LOG2E, -SCALE_LOG2E));
      ss[v] += p;
      y0[v] = fmaf(p, va, y0[v]);
      y1[v] = fmaf(p, vb, y1[v]);
    }
  }
  // merge the 16 lanes of each half-wave (pure sum/max), then write
  int bo = bh >> 1, h = bh & 1;
#pragma unroll
  for (int v = 0; v < 8; ++v) {
    float mv = mm[v], sv = ss[v], u0 = y0[v], u1 = y1[v];
#pragma unroll
    for (int mask = 1; mask <= 8; mask <<= 1) {
      mv = fmaxf(mv, __shfl_xor(mv, mask, 32));
      sv += __shfl_xor(sv, mask, 32);
      u0 += __shfl_xor(u0, mask, 32);
      u1 += __shfl_xor(u1, mask, 32);
    }
    if (l15 == v) {
      int i = qt * 16 + half * 8 + v;             // global query row
      float inv = 1.0f / sv;
      float r0 = u0 * inv, r1 = u1 * inv;
      size_t ob = (size_t)(NK + i) * 12 + bo * 6;
      out[ob + h * 2]     = r0;                   // out[4096+i, b, 2h..2h+1]
      out[ob + h * 2 + 1] = r1;
      out[ob + 4 + h]     = mv;                   // similarity
      out[SRCY_BASE + (size_t)i * 8 + bo * 4 + h * 2]     = r0;
      out[SRCY_BASE + (size_t)i * 8 + bo * 4 + h * 2 + 1] = r1;
      out[SMAP_BASE + (size_t)i * 4 + bo * 2 + h]         = mv;
    }
  }
}

extern "C" void kernel_launch(void* const* d_in, const int* in_sizes, int n_in,
                              void* d_out, int out_size, void* d_ws, size_t ws_size,
                              hipStream_t stream) {
  (void)in_sizes; (void)n_in; (void)out_size; (void)ws_size;
  const float* src     = (const float*)d_in[0];
  const float* ref_ab  = (const float*)d_in[1];
  const float* pos     = (const float*)d_in[2];
  const float* theta_w = (const float*)d_in[3];
  const float* theta_b = (const float*)d_in[4];
  const float* phi_w   = (const float*)d_in[5];
  const float* phi_b   = (const float*)d_in[6];
  const float* pos_w   = (const float*)d_in[7];
  const float* pos_b   = (const float*)d_in[8];
  float* out = (float*)d_out;

  // workspace map (floats): pos_e[1M] theta_raw[1M] phi_raw[1M] sums[512]
  // then f16 theta[1M halfs] phi[1M halfs]  -> ~16.8 MB total, 32B aligned
  float* wsf       = (float*)d_ws;
  float* pos_e     = wsf;
  float* theta_raw = wsf + 1048576;
  float* phi_raw   = wsf + 2 * 1048576;
  float* sums      = wsf + 3 * 1048576;
  _Float16* th     = (_Float16*)(wsf + 3 * 1048576 + 512);
  _Float16* ph     = th + 1048576;

  k0_fill<<<34, 256, 0, stream>>>(ref_ab, out, sums);
  k1_pos_gemm<<<512, 256, 0, stream>>>(pos, pos_w, pos_b, pos_e);
  k2_proj<<<dim3(32, 4, 2), 256, 0, stream>>>(src, pos_e, theta_w, theta_b,
                                              phi_w, phi_b, theta_raw, phi_raw, sums);
  k3_norm<<<dim3(512, 4, 2), 256, 0, stream>>>(theta_raw, phi_raw, sums, th, ph);
  k4_attn<<<dim3(32, 4), 256, 0, stream>>>(th, ph, ref_ab, out);
}
